// NeuralOoOScheduler_18975165514067
// MI455X (gfx1250) — matrix operand order repaired and checked by reading, not compiled, not run on hardware
//
#include <hip/hip_runtime.h>

// Hazard matrices as one-hot WMMA: A[i,k]=onehot(rd_i)&(rd_i<31) (16x32 f16),
// B[k,j] = (k==rn_j || k==rm_j) + 2*(k==rd_j && rd_j<31). A is one-hot, so
// D = A x B gives idx = raw + 2*waw in {0,1,2,3} exactly -> dep via a cubic
// through the 4 possible sigmoid values (3 FMAs/element). triu kills the 6
// strictly-lower 16x16 tiles -> pure b128 zero stores, 10 WMMAs per row.
// Kernel is store-BW bound: ~138 MB @ 23.3 TB/s ~= 5.9 us roofline.

typedef __attribute__((ext_vector_type(16))) _Float16 v16h;
typedef __attribute__((ext_vector_type(8)))  float    v8f;

#define WAVES_PER_BLOCK 8

__global__ void __launch_bounds__(256)
hazard_wmma_kernel(const int* __restrict__ inst,
                   const float* __restrict__ hw,
                   float* __restrict__ dep_out,
                   float* __restrict__ ready_out,
                   int B)
{
    __shared__ unsigned s_rd[WAVES_PER_BLOCK][64];
    __shared__ unsigned s_rn[WAVES_PER_BLOCK][64];
    __shared__ unsigned s_rm[WAVES_PER_BLOCK][64];

    const int wave = threadIdx.x >> 5;
    const int lane = threadIdx.x & 31;
    int b = blockIdx.x * WAVES_PER_BLOCK + wave;
    if (b >= B) b = B - 1;          // clamp (not branch) -> EXEC stays all-1s for WMMA

    // Decode two instructions per lane into LDS (rd/rn/rm fields).
#pragma unroll
    for (int t = 0; t < 2; ++t) {
        const int j = lane + 32 * t;
        const unsigned u = (unsigned)inst[(size_t)b * 64 + j];
        s_rd[wave][j] = u & 31u;
        s_rn[wave][j] = (u >> 5) & 31u;
        s_rm[wave][j] = (u >> 16) & 31u;
    }
    __syncthreads();

    // score = w0*raw + w2*waw + w4; idx = raw + 2*waw in {0,1,2,3}.
    const float w0 = hw[0], w2 = hw[2], w4 = hw[4];
    const float y0 = 1.f / (1.f + __expf(-w4));               // idx 0
    const float y1 = 1.f / (1.f + __expf(-(w0 + w4)));        // idx 1: raw
    const float y2 = 1.f / (1.f + __expf(-(w2 + w4)));        // idx 2: waw
    const float y3 = 1.f / (1.f + __expf(-(w0 + w2 + w4)));   // idx 3: both
    // Cubic through (0,y0)..(3,y3) -- exact at the 4 integer points.
    const float f01   = y1 - y0;
    const float f012  = (y2 - 2.f * y1 + y0) * 0.5f;
    const float f0123 = (y3 - 3.f * y2 + 3.f * y1 - y0) * (1.f / 6.f);
    const float a0 = y0;
    const float a1 = f01 - f012 + 2.f * f0123;
    const float a2 = f012 - 3.f * f0123;
    const float a3 = f0123;

    const int g  = lane >> 4;       // lane half (K-range selector for A/B frags)
    const int nj = lane & 15;       // M for A frag, N for B/C/D frags

    // A fragments for all 4 row-tiles, hoisted out of the tj loop.
    // A layout (16-bit 16x32): elem e -> K=16g+e (matches B fragment).
    v16h afrag[4];
#pragma unroll
    for (int ti = 0; ti < 4; ++ti) {
        const unsigned d = s_rd[wave][ti * 16 + nj];
        const int d_l = (d < 31u) ? ((int)d - 16 * g) : -1000;
#pragma unroll
        for (int e = 0; e < 16; ++e) {
            const int kk = e;   // compile-time constant
            afrag[ti][e] = (d_l == kk) ? (_Float16)1.f : (_Float16)0.f;
        }
    }

    // Per-lane store base: all tile/row/col offsets fold into store ioffset.
    float* const p0 = dep_out + (size_t)b * 4096 + (size_t)(8 * g) * 64 + (size_t)nj;

#pragma unroll
    for (int tj = 0; tj < 4; ++tj) {
        const int j = tj * 16 + nj;
        const int rn_l = (int)s_rn[wave][j] - 16 * g;
        const int rm_l = (int)s_rm[wave][j] - 16 * g;
        const unsigned rdj = s_rd[wave][j];
        const int rd_l = (rdj < 31u) ? ((int)rdj - 16 * g) : -1000;

        // Combined B fragment (32x16 f16): lane half g covers K in [16g,16g+16).
        v16h bcomb;
#pragma unroll
        for (int e = 0; e < 16; ++e) {
            const _Float16 vr = (rn_l == e || rm_l == e) ? (_Float16)1.f : (_Float16)0.f;
            const _Float16 vw = (rd_l == e) ? (_Float16)2.f : (_Float16)0.f;
            bcomb[e] = vr + vw;
        }

        float prod = 1.f;           // readiness partial over this lane's i rows
#pragma unroll
        for (int ti = 0; ti < 4; ++ti) {
            if (ti > tj) continue;  // strictly-lower tiles handled by zero pass
            v8f zacc = {};
            v8f c = __builtin_amdgcn_wmma_f32_16x16x32_f16(
                false, afrag[ti], false, bcomb, (short)0, zacc, false, false);
#pragma unroll
            for (int r = 0; r < 8; ++r) {
                const float idx = c[r];                 // exact in {0,1,2,3}
                float dep = fmaf(fmaf(fmaf(a3, idx, a2), idx, a1), idx, a0);
                if (ti == tj)                            // diagonal: triu(k=1)
                    dep = (nj > r + 8 * g) ? dep : 0.f;
                p0[ti * 1024 + r * 64 + tj * 16] = dep;  // imm-offset store
                prod *= (1.f - dep);
            }
        }
        // Combine the two lane halves (i rows r+0..7 vs r+8..15) for column j.
        prod *= __shfl_xor(prod, 16, 32);
        if (g == 0)
            ready_out[(size_t)b * 64 + (size_t)j] = prod;
    }

    // Zero-fill strictly-lower tiles (ti > tj): 6 tiles, b128 stores, no compute.
    const float4 z4 = make_float4(0.f, 0.f, 0.f, 0.f);
    float* const q0 = dep_out + (size_t)b * 4096
                    + (size_t)(lane >> 2) * 64 + (size_t)(lane & 3) * 4;
#pragma unroll
    for (int ti = 1; ti < 4; ++ti) {
#pragma unroll
        for (int tj2 = 0; tj2 < 4; ++tj2) {
            if (tj2 >= ti) continue;
            *(float4*)(q0 + ti * 1024 + tj2 * 16)       = z4;   // rows 0..7
            *(float4*)(q0 + ti * 1024 + 512 + tj2 * 16) = z4;   // rows 8..15
        }
    }
}

extern "C" void kernel_launch(void* const* d_in, const int* in_sizes, int n_in,
                              void* d_out, int out_size, void* d_ws, size_t ws_size,
                              hipStream_t stream) {
    // Input order: instructions, pc, opcode_table, feat_W, feat_b, hazard_weights.
    // pc / opcode_table / feat_W / feat_b feed only the deleted `emb` -> unused.
    const int*   inst = (const int*)d_in[0];
    const float* hw   = (const float*)d_in[5];

    const int B = in_sizes[0] / 64;                 // W fixed at 64
    float* dep   = (float*)d_out;                   // (B, 64, 64)
    float* ready = dep + (size_t)B * 64 * 64;       // (B, 64)

    dim3 block(256);
    dim3 grid((B + WAVES_PER_BLOCK - 1) / WAVES_PER_BLOCK);
    hazard_wmma_kernel<<<grid, block, 0, stream>>>(inst, hw, dep, ready, B);
}
